// FluxAttention_34866544509146
// MI455X (gfx1250) — matrix operand order repaired and checked by reading, not compiled, not run on hardware
//
#include <hip/hip_runtime.h>
#include <hip/hip_bf16.h>

typedef __bf16 bf16_t;
typedef __attribute__((ext_vector_type(16))) __bf16 v16bf;
typedef __attribute__((ext_vector_type(8)))  float  v8f;

#define S_TXT_N 512
#define S_IMG_N 2048
#define S_ALL_N 2560
#define CDIM    3072
#define NHEAD   24
#define HDIM    128

__device__ __forceinline__ v8f wmma_bf16(const v16bf& a, const v16bf& b, const v8f& c) {
    return __builtin_amdgcn_wmma_f32_16x16x32_bf16(false, a, false, b, (short)0, c, false, false);
}

// 16 contiguous bf16 (32B) -> fragment. p must be 16B aligned.
__device__ __forceinline__ v16bf load_frag_contig(const bf16_t* p) {
    union { v16bf v; uint4 u[2]; } r;
    r.u[0] = *(const uint4*)p;
    r.u[1] = *(const uint4*)(p + 8);
    return r.v;
}
// A-fragment: two 16B chunks at +0 and +16 elements (ISA K split 0-7 / 16-23).
__device__ __forceinline__ v16bf load_frag_a(const bf16_t* p) {
    union { v16bf v; uint4 u[2]; } r;
    r.u[0] = *(const uint4*)p;
    r.u[1] = *(const uint4*)(p + 16);
    return r.v;
}

// ---------------------------------------------------------------------------
// Generic GEMM:  C[M,N] = A[M,K] (f32) * B[K,N] (f32) + bias[N], f32 out.
// f32 -> bf16 conversion while staging into LDS; B staged TRANSPOSED so that
// every B fragment is one contiguous 32B LDS read.
// Workgroup tile 128x128, BK=32. 8 waves (4 M x 2 N), each wave 32x64.
// ---------------------------------------------------------------------------
__global__ __launch_bounds__(256) void gemm_bias_kernel(
    const float* __restrict__ A, const float* __restrict__ B,
    const float* __restrict__ bias, float* __restrict__ C,
    int M, int N, int K)
{
    constexpr int BM = 128, BN = 128, BK = 32;
    constexpr int ASTR = BK + 8;   // 40: row stride, 80B -> 16B aligned chunks
    constexpr int BTSTR = BK + 8;  // 40: Bt col stride (Bt[col][k])
    __shared__ bf16_t As[BM * ASTR];
    __shared__ bf16_t Bt[BN * BTSTR];

    const int tid  = threadIdx.x;
    const int lane = tid & 31;
    const int wave = tid >> 5;
    const int wm   = wave >> 1;        // 0..3
    const int wn   = wave & 1;         // 0..1
    const int l16  = lane & 15;
    const int half = lane >> 4;        // lane-half (ISA K/M split)

    const int bm = blockIdx.y * BM;
    const int bn = blockIdx.x * BN;

    v8f acc[2][4];
    #pragma unroll
    for (int i = 0; i < 2; ++i)
        #pragma unroll
        for (int j = 0; j < 4; ++j) { v8f z = {}; acc[i][j] = z; }

    for (int kk = 0; kk < K; kk += BK) {
        // Stage A tile (BM x BK): 1024 float4 / 256 threads
        #pragma unroll
        for (int it = 0; it < 4; ++it) {
            int f4 = tid + 256 * it;
            int r  = f4 >> 3;
            int c4 = (f4 & 7) * 4;
            const float4 v = *(const float4*)(A + (size_t)(bm + r) * K + kk + c4);
            bf16_t* dst = &As[r * ASTR + c4];
            dst[0] = (bf16_t)v.x; dst[1] = (bf16_t)v.y;
            dst[2] = (bf16_t)v.z; dst[3] = (bf16_t)v.w;
        }
        // Stage B tile (BK x BN) transposed into Bt[col][k]
        #pragma unroll
        for (int it = 0; it < 4; ++it) {
            int f4 = tid + 256 * it;
            int r  = f4 >> 5;              // k row 0..31
            int c4 = (f4 & 31) * 4;        // col
            const float4 v = *(const float4*)(B + (size_t)(kk + r) * N + bn + c4);
            Bt[(c4 + 0) * BTSTR + r] = (bf16_t)v.x;
            Bt[(c4 + 1) * BTSTR + r] = (bf16_t)v.y;
            Bt[(c4 + 2) * BTSTR + r] = (bf16_t)v.z;
            Bt[(c4 + 3) * BTSTR + r] = (bf16_t)v.w;
        }
        __syncthreads();

        // B fragments: contiguous 32B per lane from transposed tile
        v16bf bfrag[4];
        #pragma unroll
        for (int sn = 0; sn < 4; ++sn) {
            const int col = wn * 64 + sn * 16 + l16;
            bfrag[sn] = load_frag_contig(&Bt[col * BTSTR + half * 16]);
        }
        #pragma unroll
        for (int sm = 0; sm < 2; ++sm) {
            const int row = wm * 32 + sm * 16 + l16;
            const v16bf afrag = load_frag_contig(&As[row * ASTR + half * 16]);
            #pragma unroll
            for (int sn = 0; sn < 4; ++sn)
                acc[sm][sn] = wmma_bf16(afrag, bfrag[sn], acc[sm][sn]);
        }
        __syncthreads();
    }

    // Epilogue: C layout — lane = col, VGPR r = row r (half 0) / r+8 (half 1)
    #pragma unroll
    for (int sm = 0; sm < 2; ++sm) {
        #pragma unroll
        for (int sn = 0; sn < 4; ++sn) {
            const int col = bn + wn * 64 + sn * 16 + l16;
            const float bv = bias ? bias[col] : 0.0f;
            #pragma unroll
            for (int r = 0; r < 8; ++r) {
                const int row = bm + wm * 32 + sm * 16 + r + half * 8;
                C[(size_t)row * N + col] = acc[sm][sn][r] + bv;
            }
        }
    }
}

// ---------------------------------------------------------------------------
// Per-(token, head) RMSNorm + rotary on q/k; emits bf16 Q/K/V [S_ALL, C].
// ---------------------------------------------------------------------------
__global__ __launch_bounds__(128) void normrope_kernel(
    const float* __restrict__ qkv,
    const float* __restrict__ img_cos, const float* __restrict__ img_sin,
    const float* __restrict__ txt_cos, const float* __restrict__ txt_sin,
    const float* __restrict__ img_q_scale, const float* __restrict__ img_k_scale,
    const float* __restrict__ txt_q_scale, const float* __restrict__ txt_k_scale,
    bf16_t* __restrict__ Qb, bf16_t* __restrict__ Kb, bf16_t* __restrict__ Vb)
{
    const int s = blockIdx.x;       // 0..2559 (txt first, then img)
    const int h = blockIdx.y;       // 0..23
    const int d = threadIdx.x;      // 0..127
    const bool is_txt = (s < S_TXT_N);
    const int  srow   = is_txt ? s : (s - S_TXT_N);
    const float* cosr = (is_txt ? txt_cos : img_cos) + (size_t)srow * HDIM;
    const float* sinr = (is_txt ? txt_sin : img_sin) + (size_t)srow * HDIM;
    const float qsc = (is_txt ? txt_q_scale : img_q_scale)[d];
    const float ksc = (is_txt ? txt_k_scale : img_k_scale)[d];

    const size_t base = (size_t)s * (3 * CDIM) + (size_t)h * HDIM + d;
    const float qv = qkv[base];
    const float kv = qkv[base + CDIM];
    const float vv = qkv[base + 2 * CDIM];

    __shared__ float red[HDIM];
    __shared__ float qsh[HDIM];
    __shared__ float ksh[HDIM];

    red[d] = qv * qv;
    __syncthreads();
    for (int o = 64; o > 0; o >>= 1) { if (d < o) red[d] += red[d + o]; __syncthreads(); }
    const float rq = rsqrtf(red[0] / (float)HDIM + 1e-6f);
    __syncthreads();

    red[d] = kv * kv;
    __syncthreads();
    for (int o = 64; o > 0; o >>= 1) { if (d < o) red[d] += red[d + o]; __syncthreads(); }
    const float rk = rsqrtf(red[0] / (float)HDIM + 1e-6f);
    __syncthreads();

    const float qn = qv * rq * qsc;
    const float kn = kv * rk * ksc;
    qsh[d] = qn; ksh[d] = kn;
    __syncthreads();

    float qo, ko;
    if ((d & 1) == 0) {  // even: x*cos - x[odd]*sin
        qo = qn * cosr[d] - qsh[d + 1] * sinr[d];
        ko = kn * cosr[d] - ksh[d + 1] * sinr[d];
    } else {             // odd:  x*cos + x[even]*sin
        qo = qn * cosr[d] + qsh[d - 1] * sinr[d];
        ko = kn * cosr[d] + ksh[d - 1] * sinr[d];
    }

    const size_t ob = (size_t)s * CDIM + (size_t)h * HDIM + d;
    Qb[ob] = (bf16_t)qo;
    Kb[ob] = (bf16_t)ko;
    Vb[ob] = (bf16_t)vv;
}

// ---------------------------------------------------------------------------
// Flash attention: block = (128 queries, 1 head). Key tiles of 32.
// Wave w owns query rows w*16..w*16+15. K tile row-major (contiguous B-frags
// for Q*K^T); V tile staged transposed (contiguous B-frags for P*V).
// ---------------------------------------------------------------------------
__global__ __launch_bounds__(256) void flash_attn_kernel(
    const bf16_t* __restrict__ Qb, const bf16_t* __restrict__ Kb,
    const bf16_t* __restrict__ Vb, float* __restrict__ Out)
{
    constexpr int QT = 128, KT = 32;
    constexpr int QSTR = HDIM + 8;     // 136 (272B rows, 16B-aligned chunks)
    constexpr int KSTR = HDIM + 8;
    constexpr int VTSTR = KT + 8;      // 40: Vt[d][key]
    constexpr int PSTR = KT + 8;       // 40: Ps[qrow][key]
    __shared__ bf16_t Qs[QT * QSTR];
    __shared__ bf16_t Ks[KT * KSTR];
    __shared__ bf16_t Vt[HDIM * VTSTR];
    __shared__ bf16_t Ps[8 * 16 * PSTR];

    const int tid = threadIdx.x, lane = tid & 31, wave = tid >> 5;
    const int l16 = lane & 15, half = lane >> 4;
    const int h = blockIdx.y;
    const int qbase = blockIdx.x * QT;
    const size_t colbase = (size_t)h * HDIM;

    // Load Q tile (128x128 bf16) with 16B vector loads
    #pragma unroll
    for (int it = 0; it < 8; ++it) {
        const int e8 = tid + 256 * it;
        const int r = e8 >> 4, c = (e8 & 15) * 8;
        *(uint4*)&Qs[r * QSTR + c] =
            *(const uint4*)&Qb[(size_t)(qbase + r) * CDIM + colbase + c];
    }
    __syncthreads();

    v8f o[8];
    #pragma unroll
    for (int n = 0; n < 8; ++n) { v8f z = {}; o[n] = z; }
    float mrow[8], lrow[8];
    #pragma unroll
    for (int r = 0; r < 8; ++r) { mrow[r] = -1e30f; lrow[r] = 0.0f; }
    const float sm_scale = 0.08838834764831845f;   // 1/sqrt(128)

    for (int kt = 0; kt < S_ALL_N; kt += KT) {
        #pragma unroll
        for (int it = 0; it < 2; ++it) {
            const int e8 = tid + 256 * it;      // 0..511
            const int r = e8 >> 4, c = (e8 & 15) * 8;
            // K tile row-major copy
            *(uint4*)&Ks[r * KSTR + c] =
                *(const uint4*)&Kb[(size_t)(kt + r) * CDIM + colbase + c];
            // V tile transposed: Vt[d][key]
            union { uint4 u; bf16_t e[8]; } vv;
            vv.u = *(const uint4*)&Vb[(size_t)(kt + r) * CDIM + colbase + c];
            #pragma unroll
            for (int i = 0; i < 8; ++i)
                Vt[(c + i) * VTSTR + r] = vv.e[i];
        }
        __syncthreads();

        // logits: s0 = keys 0-15, s1 = keys 16-31 ; K-dim = 128 = 4 x 32
        v8f s0 = {}, s1 = {};
        const int qrow = wave * 16 + l16;
        #pragma unroll
        for (int kc = 0; kc < 4; ++kc) {
            const v16bf aq = load_frag_contig(&Qs[qrow * QSTR + kc * 32 + half * 16]);
            // B[kd][n] = K[key n][kd]: row-major Ks rows are contiguous in kd
            const v16bf bk0 = load_frag_contig(&Ks[l16 * KSTR + kc * 32 + half * 16]);
            const v16bf bk1 = load_frag_contig(&Ks[(16 + l16) * KSTR + kc * 32 + half * 16]);
            s0 = wmma_bf16(aq, bk0, s0);
            s1 = wmma_bf16(aq, bk1, s1);
        }

        // online softmax (per row r, actual row = r + half*8)
        #pragma unroll
        for (int r = 0; r < 8; ++r) {
            const float x0 = s0[r] * sm_scale;
            const float x1 = s1[r] * sm_scale;
            float mt = fmaxf(x0, x1);
            #pragma unroll
            for (int msk = 1; msk < 16; msk <<= 1) mt = fmaxf(mt, __shfl_xor(mt, msk));
            const float mn = fmaxf(mrow[r], mt);
            const float corr = __expf(mrow[r] - mn);
            mrow[r] = mn;
            const float p0 = __expf(x0 - mn);
            const float p1 = __expf(x1 - mn);
            float rs = p0 + p1;
            #pragma unroll
            for (int msk = 1; msk < 16; msk <<= 1) rs += __shfl_xor(rs, msk);
            lrow[r] = lrow[r] * corr + rs;
            #pragma unroll
            for (int n = 0; n < 8; ++n) o[n][r] *= corr;
            const int prow = wave * 16 + r + half * 8;
            Ps[prow * PSTR + l16]      = (bf16_t)p0;
            Ps[prow * PSTR + 16 + l16] = (bf16_t)p1;
        }
        // intra-wave LDS RAW: stores above feed A-fragment loads below
        asm volatile("s_wait_dscnt 0" ::: "memory");

        // O += P(16x32) @ V(32x128)
        const v16bf ap = load_frag_contig(&Ps[(wave * 16 + l16) * PSTR + half * 16]);
        #pragma unroll
        for (int n = 0; n < 8; ++n) {
            const int col = n * 16 + l16;   // d index
            const v16bf bv = load_frag_contig(&Vt[col * VTSTR + half * 16]);
            o[n] = wmma_bf16(ap, bv, o[n]);
        }
        __syncthreads();   // protect Ks/Vt before next tile load
    }

    // normalize + store attention output (f32, [S_ALL, C])
    #pragma unroll
    for (int n = 0; n < 8; ++n) {
        const int col = n * 16 + l16;
        #pragma unroll
        for (int r = 0; r < 8; ++r) {
            const int row = qbase + wave * 16 + r + half * 8;
            Out[(size_t)row * CDIM + colbase + col] = o[n][r] / lrow[r];
        }
    }
}

// ---------------------------------------------------------------------------
extern "C" void kernel_launch(void* const* d_in, const int* in_sizes, int n_in,
                              void* d_out, int out_size, void* d_ws, size_t ws_size,
                              hipStream_t stream) {
    const float* hidden   = (const float*)d_in[0];
    const float* enc      = (const float*)d_in[1];
    const float* img_cos  = (const float*)d_in[2];
    const float* img_sin  = (const float*)d_in[3];
    const float* txt_cos  = (const float*)d_in[4];
    const float* txt_sin  = (const float*)d_in[5];
    const float* img_qkv_w = (const float*)d_in[6];
    const float* img_qkv_b = (const float*)d_in[7];
    const float* img_proj_w = (const float*)d_in[8];
    const float* img_proj_b = (const float*)d_in[9];
    const float* img_q_scale = (const float*)d_in[10];
    const float* img_k_scale = (const float*)d_in[11];
    const float* txt_qkv_w = (const float*)d_in[12];
    const float* txt_qkv_b = (const float*)d_in[13];
    const float* txt_proj_w = (const float*)d_in[14];
    const float* txt_proj_b = (const float*)d_in[15];
    const float* txt_q_scale = (const float*)d_in[16];
    const float* txt_k_scale = (const float*)d_in[17];

    char* ws = (char*)d_ws;
    size_t off = 0;
    float* qkv = (float*)(ws + off);  off += (size_t)S_ALL_N * 3 * CDIM * sizeof(float);
    bf16_t* Qb = (bf16_t*)(ws + off); off += (size_t)S_ALL_N * CDIM * sizeof(bf16_t);
    bf16_t* Kb = (bf16_t*)(ws + off); off += (size_t)S_ALL_N * CDIM * sizeof(bf16_t);
    bf16_t* Vb = (bf16_t*)(ws + off); off += (size_t)S_ALL_N * CDIM * sizeof(bf16_t);
    float* attn = (float*)(ws + off); off += (size_t)S_ALL_N * CDIM * sizeof(float);
    (void)ws_size; (void)in_sizes; (void)n_in; (void)out_size;

    const dim3 blk(256);

    // QKV projections (txt rows 0..511, img rows 512..2559 -> concat order)
    gemm_bias_kernel<<<dim3(3 * CDIM / 128, S_TXT_N / 128), blk, 0, stream>>>(
        enc, txt_qkv_w, txt_qkv_b, qkv, S_TXT_N, 3 * CDIM, CDIM);
    gemm_bias_kernel<<<dim3(3 * CDIM / 128, S_IMG_N / 128), blk, 0, stream>>>(
        hidden, img_qkv_w, img_qkv_b, qkv + (size_t)S_TXT_N * 3 * CDIM,
        S_IMG_N, 3 * CDIM, CDIM);

    // RMSNorm + rotary -> bf16 Q/K/V
    normrope_kernel<<<dim3(S_ALL_N, NHEAD), dim3(128), 0, stream>>>(
        qkv, img_cos, img_sin, txt_cos, txt_sin,
        img_q_scale, img_k_scale, txt_q_scale, txt_k_scale, Qb, Kb, Vb);

    // Flash attention
    flash_attn_kernel<<<dim3(S_ALL_N / 128, NHEAD), blk, 0, stream>>>(Qb, Kb, Vb, attn);

    // Output projections: d_out = [img_out (2048x3072) | txt_out (512x3072)]
    float* out = (float*)d_out;
    gemm_bias_kernel<<<dim3(CDIM / 128, S_IMG_N / 128), blk, 0, stream>>>(
        attn + (size_t)S_TXT_N * CDIM, img_proj_w, img_proj_b, out,
        S_IMG_N, CDIM, CDIM);
    gemm_bias_kernel<<<dim3(CDIM / 128, S_TXT_N / 128), blk, 0, stream>>>(
        attn, txt_proj_w, txt_proj_b, out + (size_t)S_IMG_N * CDIM,
        S_TXT_N, CDIM, CDIM);
}
